// TFLiteISTFT_15865609191758
// MI455X (gfx1250) — matrix operand order repaired and checked by reading, not compiled, not run on hardware
//
#include <hip/hip_runtime.h>

// ---------------------------------------------------------------------------
// ISTFT as GEMM + overlap-add for MI455X (gfx1250, wave32, WMMA bf16).
//
//   frames[bt, n] = sum_{k=0..512} re[bt,k]*Wre[k,n] + im[bt,k]*Wim[k,n]
//   Wre[k,n] = (cos[n,k] + (1<=k<=511 ? cos[n,1024-k] : 0)) * win[n]/1024
//   Wim[k,n] = (-sin[n,k] + (1<=k<=511 ? sin[n,1024-k] : 0)) * win[n]/1024
//   out[b, t*256 + n] += frames[b*2000+t, n]
//
// GEMM: M=32000, N=1024, K=1026 (re 0..512, im 513..1025), K padded to 1056.
// bf16 hi/lo split (3 WMMA terms: hh + hl + lh) ~ 17-bit mantissa accuracy.
// Split done once per tile during LDS staging (v_and + v_sub + v_perm).
//
// Tiling: block = 256 threads = 8 waves (4x2), block tile 256x128,
// wave tile 64x64 = 4x4 fragments -> 48 v_wmma per wave per K-step.
// ---------------------------------------------------------------------------

typedef __attribute__((ext_vector_type(16))) __bf16          v16bf;
typedef __attribute__((ext_vector_type(8)))  float           v8f;
typedef __attribute__((ext_vector_type(8)))  unsigned short  us8;

#define BATCH   16
#define TT      2000
#define BINS    513
#define NFFT    1024
#define STEP    256
#define OUTLEN  512768          // (2000-1)*256 + 1024
#define MROWS   (BATCH * TT)    // 32000
#define KDIM    1026            // 513 re + 513 im
#define KSTEPS  33              // K padded to 1056
#define NTILES  64              // 1024 / 16
#define WELEMS  (KSTEPS * NTILES * 32 * 16)   // packed weight elements (bf16)
#define ASTRIDE 40              // LDS row stride in ushorts (80B, 16B-aligned)
#define MTILE   256             // block M tile
#define NPAIRS  (MTILE * 16)    // staging pairs per K-step

__device__ __forceinline__ unsigned short bf16_rne(float f) {
    unsigned int u = __float_as_uint(f);
    u += 0x7FFFu + ((u >> 16) & 1u);
    return (unsigned short)(u >> 16);
}
__device__ __forceinline__ float bf16_to_f32(unsigned short h) {
    return __uint_as_float((unsigned int)h << 16);
}

// ---------------------------------------------------------------------------
// Pack folded IDFT weights into per-lane WMMA B-fragment order (bf16 hi/lo).
// Layout: whi[((ks*64 + nt)*32 + lane)*16 + j]
//   lane holds column n = nt*16 + lane%16, K offsets (lane/16)*16 + j
// contiguous K order (not the ISA table order); A fragments use the same order.
// ---------------------------------------------------------------------------
__global__ void pack_weights(const float* __restrict__ idft_cos,
                             const float* __restrict__ idft_sin,
                             const float* __restrict__ window,
                             unsigned short* __restrict__ whi,
                             unsigned short* __restrict__ wlo) {
    int id = blockIdx.x * 256 + threadIdx.x;
    if (id >= WELEMS) return;
    int j    = id & 15;
    int lane = (id >> 4) & 31;
    int nt   = (id >> 9) & 63;
    int ks   = id >> 15;
    int k = ks * 32 + (lane >> 4) * 16 + j;   // logical K row (0..1055)
    int n = nt * 16 + (lane & 15);            // output sample within frame

    float v = 0.0f;
    if (k < KDIM) {
        float ws = window[n] * (1.0f / 1024.0f);
        if (k < BINS) {                       // real part rows
            float c = idft_cos[n * NFFT + k];
            if (k >= 1 && k <= 511) c += idft_cos[n * NFFT + (NFFT - k)];
            v = c * ws;
        } else {                              // imag part rows
            int kk = k - BINS;
            float s = -idft_sin[n * NFFT + kk];
            if (kk >= 1 && kk <= 511) s += idft_sin[n * NFFT + (NFFT - kk)];
            v = s * ws;
        }
    }
    unsigned short h = bf16_rne(v);
    whi[id] = h;
    wlo[id] = bf16_rne(v - bf16_to_f32(h));
}

__global__ void zero_out(float* __restrict__ out, int n) {
    int i = blockIdx.x * 256 + threadIdx.x;
    if (i < n) out[i] = 0.0f;
}

// ---------------------------------------------------------------------------
// GEMM + overlap-add. Grid (125, 8): 256x128 output tile per block.
// 8 waves arranged 4x2; each wave owns a 64x64 sub-tile = 4x4 fragments.
// LDS holds the A tile pre-split into bf16 hi/lo planes.
// ---------------------------------------------------------------------------
__launch_bounds__(256)
__global__ void istft_wmma(const float* __restrict__ re,
                           const float* __restrict__ im,
                           const unsigned short* __restrict__ whi,
                           const unsigned short* __restrict__ wlo,
                           float* __restrict__ out) {
    __shared__ unsigned short Ah[MTILE * ASTRIDE];   // 20,480 B
    __shared__ unsigned short Al[MTILE * ASTRIDE];   // 20,480 B

    const int tid     = threadIdx.x;
    const int lane    = tid & 31;
    const int wave    = tid >> 5;
    const int wm      = wave >> 1;            // 0..3  (M quarter)
    const int wn      = wave & 1;             // 0..1  (N half)
    const int lane_n  = lane & 15;            // col within fragment / A row
    const int lane_hi = lane >> 4;            // 0/1
    const int kbase   = lane_hi * 16;         // A-fragment K sub-base
    const int m0      = blockIdx.x * MTILE;
    const int nt0     = blockIdx.y * 8 + wn * 4;

    v8f acc[4][4] = {};

    for (int ks = 0; ks < KSTEPS; ++ks) {
        const int k0 = ks * 32;

        // ---- stage A tile (256 rows x 32 K) into LDS as bf16 hi/lo planes.
        // Each thread handles 16 pairs of adjacent K values: load 2 f32,
        // truncation-split, pack two bf16 per dword with v_perm_b32.
        #pragma unroll
        for (int i = 0; i < NPAIRS / 256; ++i) {
            int p   = tid + (i << 8);         // pair index 0..4095
            int r   = p >> 4;                 // row 0..255
            int kp  = (p & 15) << 1;          // even K offset 0..30
            int row = m0 + r;
            int k   = k0 + kp;
            float x0, x1;
            if (k0 + 31 < BINS) {             // pure real window (ks <= 15)
                x0 = re[row * BINS + k];
                x1 = re[row * BINS + k + 1];
            } else if (k0 >= BINS && k0 + 31 < KDIM) {  // pure imag (17..31)
                x0 = im[row * BINS + (k - BINS)];
                x1 = im[row * BINS + (k - BINS) + 1];
            } else {                          // boundary steps ks==16, ks==32
                x0 = 0.0f; x1 = 0.0f;
                if (k < BINS)          x0 = re[row * BINS + k];
                else if (k < KDIM)     x0 = im[row * BINS + (k - BINS)];
                int k1 = k + 1;
                if (k1 < BINS)         x1 = re[row * BINS + k1];
                else if (k1 < KDIM)    x1 = im[row * BINS + (k1 - BINS)];
            }
            unsigned int u0 = __float_as_uint(x0);
            unsigned int u1 = __float_as_uint(x1);
            unsigned int h0 = u0 & 0xFFFF0000u;
            unsigned int h1 = u1 & 0xFFFF0000u;
            float l0 = x0 - __uint_as_float(h0);   // exact residual
            float l1 = x1 - __uint_as_float(h1);
            // pack [top16(u0), top16(u1)] -> one dword (v_perm_b32)
            unsigned int hp = __builtin_amdgcn_perm(u1, u0, 0x07060302u);
            unsigned int lp = __builtin_amdgcn_perm(__float_as_uint(l1),
                                                    __float_as_uint(l0),
                                                    0x07060302u);
            *(unsigned int*)&Ah[r * ASTRIDE + kp] = hp;
            *(unsigned int*)&Al[r * ASTRIDE + kp] = lp;
        }
        __syncthreads();

        // ---- B fragments: contiguous 32B per lane from packed weights
        v16bf bhi[4], blo[4];
        #pragma unroll
        for (int ni = 0; ni < 4; ++ni) {
            size_t off = ((size_t)((ks * 64 + nt0 + ni) * 32 + lane)) * 16;
            bhi[ni] = *(const v16bf*)(whi + off);
            blo[ni] = *(const v16bf*)(wlo + off);
        }

        // ---- per M-fragment: 2 ds_load_b128 (hi) + 2 (lo), then 12 WMMAs
        // A layout: row = lane%16, K = kbase..kbase+15 (contiguous, matches B)
        #pragma unroll
        for (int mi = 0; mi < 4; ++mi) {
            const int r = wm * 64 + mi * 16 + lane_n;
            const unsigned short* ph = &Ah[r * ASTRIDE + kbase];
            const unsigned short* pl = &Al[r * ASTRIDE + kbase];
            union { v16bf v; us8 h[2]; } H, L;
            H.h[0] = *(const us8*)(ph);
            H.h[1] = *(const us8*)(ph + 8);
            L.h[0] = *(const us8*)(pl);
            L.h[1] = *(const us8*)(pl + 8);
            v16bf ahi = H.v, alo = L.v;
            #pragma unroll
            for (int ni = 0; ni < 4; ++ni) {
                acc[mi][ni] = __builtin_amdgcn_wmma_f32_16x16x32_bf16(
                    false, ahi, false, bhi[ni], (short)0, acc[mi][ni], false, false);
                acc[mi][ni] = __builtin_amdgcn_wmma_f32_16x16x32_bf16(
                    false, ahi, false, blo[ni], (short)0, acc[mi][ni], false, false);
                acc[mi][ni] = __builtin_amdgcn_wmma_f32_16x16x32_bf16(
                    false, alo, false, bhi[ni], (short)0, acc[mi][ni], false, false);
            }
        }
        __syncthreads();
    }

    // ---- epilogue: overlap-add  out[b, t*256 + n] += D[m, n]
    // D layout: lane holds n = frag_n + lane%16; vgpr vr holds m = frag_m + vr + 8*(lane/16)
    #pragma unroll
    for (int mi = 0; mi < 4; ++mi) {
        const int mbase = m0 + wm * 64 + mi * 16 + lane_hi * 8;
        #pragma unroll
        for (int ni = 0; ni < 4; ++ni) {
            const int n = blockIdx.y * 128 + wn * 64 + ni * 16 + lane_n;
            #pragma unroll
            for (int vr = 0; vr < 8; ++vr) {
                int m = mbase + vr;
                int b = m / TT;
                int t = m - b * TT;
                atomicAdd(out + ((size_t)b * OUTLEN + t * STEP + n), acc[mi][ni][vr]);
            }
        }
    }
}

// ---------------------------------------------------------------------------
extern "C" void kernel_launch(void* const* d_in, const int* in_sizes, int n_in,
                              void* d_out, int out_size, void* d_ws, size_t ws_size,
                              hipStream_t stream) {
    const float* stft_real = (const float*)d_in[0];
    const float* stft_imag = (const float*)d_in[1];
    const float* idft_cos  = (const float*)d_in[2];
    const float* idft_sin  = (const float*)d_in[3];
    const float* window    = (const float*)d_in[4];
    float* out = (float*)d_out;

    unsigned short* whi = (unsigned short*)d_ws;       // 2.16 MB
    unsigned short* wlo = whi + WELEMS;                // 2.16 MB (4.3 MB total ws)

    pack_weights<<<(WELEMS + 255) / 256, 256, 0, stream>>>(idft_cos, idft_sin,
                                                           window, whi, wlo);
    zero_out<<<(out_size + 255) / 256, 256, 0, stream>>>(out, out_size);
    istft_wmma<<<dim3(MROWS / MTILE, NFFT / 128), 256, 0, stream>>>(
        stft_real, stft_imag, whi, wlo, out);
}
